// BitLinear_89739046683230
// MI455X (gfx1250) — compile-verified, matrix operand order checked by reading, not run on hardware
//
#include <hip/hip_runtime.h>
#include <hip/hip_bf16.h>
#include <hip/hip_fp16.h>

// ---------------------------------------------------------------------------
// BitLinear forward for MI455X (gfx1250, wave32, WMMA f16 16x16x32)
//   out = (x_q @ w_q) * beta * gamma_row
// w_q = +/-1 exact in f16; x_q in (-1,1) -> f16 (~5e-4 rel), f32 WMMA accum.
// HBM-bound (~390MB @ 23.3TB/s ~ 17us); 64x64 wave tile -> 32 FLOP/B from L2.
// ---------------------------------------------------------------------------

typedef __attribute__((ext_vector_type(16))) _Float16 v16h;
typedef __attribute__((ext_vector_type(8)))  _Float16 v8h;
typedef __attribute__((ext_vector_type(4)))  _Float16 v4h;
typedef __attribute__((ext_vector_type(8)))  float    v8f;

#define N_ROWS   32768
#define K_DIM    1024
#define N_OUT    1024
#define EPSF     1e-5f

// workspace layout (bytes)
#define WS_SCALARS 0                         // [0]=sum(f32 bits), [1]=absmax bits
#define WS_GAMMA   256                       // 32768 floats
#define WS_XQ      (256 + N_ROWS * 4)        // 32768*1024 f16 (64 MB), 256B aligned
#define WS_WQ      (WS_XQ + (size_t)N_ROWS * K_DIM * 2)  // 1024*1024 f16 (2 MB)

// ---------------------------------------------------------------------------
__global__ void bl_init_kernel(unsigned* scalars) {
    if (threadIdx.x == 0) { scalars[0] = 0u; scalars[1] = 0u; }
}

// ---------------------------------------------------------------------------
// Reduce weight: sum (for mean) and max|w| (beta). 1024 blocks x 256 thr x 4 elem.
__global__ __launch_bounds__(256) void bl_wreduce_kernel(const float* __restrict__ w,
                                                         unsigned* __restrict__ scalars) {
    __shared__ float s_sum[256];
    __shared__ float s_max[256];
    const int tid = threadIdx.x;
    const size_t base = ((size_t)blockIdx.x * 256 + tid) * 4;
    const float4 v = *(const float4*)(w + base);
    float sum = v.x + v.y + v.z + v.w;
    float mx  = fmaxf(fmaxf(fabsf(v.x), fabsf(v.y)), fmaxf(fabsf(v.z), fabsf(v.w)));
    s_sum[tid] = sum; s_max[tid] = mx;
    __syncthreads();
    for (int s = 128; s > 0; s >>= 1) {
        if (tid < s) {
            s_sum[tid] += s_sum[tid + s];
            s_max[tid] = fmaxf(s_max[tid], s_max[tid + s]);
        }
        __syncthreads();
    }
    if (tid == 0) {
        atomicAdd((float*)&scalars[0], s_sum[0]);
        atomicMax(&scalars[1], __float_as_uint(s_max[0]));  // abs-bits monotone as uint
    }
}

// ---------------------------------------------------------------------------
// w_q[t] transpose: wq_t[n][k] = sign(w[k][n] - mean), f16. LDS 32x32 tile.
__global__ __launch_bounds__(256) void bl_wquant_kernel(const float* __restrict__ w,
                                                        const unsigned* __restrict__ scalars,
                                                        _Float16* __restrict__ wq_t) {
    __shared__ _Float16 tile[32][33];
    const float a = __uint_as_float(*(const volatile unsigned*)&scalars[0]) * (1.0f / (K_DIM * N_OUT));
    const int tx = threadIdx.x & 31;   // fast dim
    const int ty = threadIdx.x >> 5;   // 0..7
    const int k0 = blockIdx.y * 32;
    const int n0 = blockIdx.x * 32;
#pragma unroll
    for (int i = 0; i < 4; ++i) {
        const int k = ty + i * 8;
        const float wv = w[(size_t)(k0 + k) * N_OUT + n0 + tx];
        const float s = (wv > a) ? 1.0f : ((wv < a) ? -1.0f : 0.0f);
        tile[k][tx] = (_Float16)s;
    }
    __syncthreads();
#pragma unroll
    for (int i = 0; i < 4; ++i) {
        const int n = ty + i * 8;
        wq_t[(size_t)(n0 + n) * K_DIM + k0 + tx] = tile[tx][n];
    }
}

// ---------------------------------------------------------------------------
// Per-row gamma = max|x|, then x_q = clip(x/(gamma+eps), +-(1-eps)) -> f16.
// One block (256 thr) per row; 4 floats/thread.
__global__ __launch_bounds__(256) void bl_xquant_kernel(const float* __restrict__ x,
                                                        float* __restrict__ gamma,
                                                        _Float16* __restrict__ xq) {
    __shared__ float s_max[256];
    const int row = blockIdx.x;
    const int tid = threadIdx.x;
    const float4 v = *(const float4*)(x + (size_t)row * K_DIM + tid * 4);
    float mx = fmaxf(fmaxf(fabsf(v.x), fabsf(v.y)), fmaxf(fabsf(v.z), fabsf(v.w)));
    s_max[tid] = mx;
    __syncthreads();
    for (int s = 128; s > 0; s >>= 1) {
        if (tid < s) s_max[tid] = fmaxf(s_max[tid], s_max[tid + s]);
        __syncthreads();
    }
    const float g = s_max[0];
    if (tid == 0) gamma[row] = g;
    const float inv = 1.0f / (g + EPSF);
    const float lim = 1.0f - EPSF;
    v4h q;
    q.x = (_Float16)fminf(fmaxf(v.x * inv, -lim), lim);
    q.y = (_Float16)fminf(fmaxf(v.y * inv, -lim), lim);
    q.z = (_Float16)fminf(fmaxf(v.z * inv, -lim), lim);
    q.w = (_Float16)fminf(fmaxf(v.w * inv, -lim), lim);
    *(v4h*)(xq + (size_t)row * K_DIM + tid * 4) = q;
}

// ---------------------------------------------------------------------------
// Load one 16-bit A/B WMMA fragment (ISA 7.12.2 layout): lane l holds row l%16,
// 8 contiguous f16 at K-offset (l/16)*8 and 8 more at +16. Two b128 loads.
__device__ __forceinline__ v16h bl_load_frag(const _Float16* p) {
    v8h lo = *(const v8h*)(p);
    v8h hi = *(const v8h*)(p + 16);
    return __builtin_shufflevector(lo, hi, 0, 1, 2, 3, 4, 5, 6, 7,
                                   8, 9, 10, 11, 12, 13, 14, 15);
}

// ---------------------------------------------------------------------------
// GEMM: block = 8 waves (256 thr) arranged 2(M) x 4(N) -> block tile 128x256.
// Each wave: 64x64 output = 4x4 tiles of 16x16 -> 16 WMMAs per K-step of 32.
// Arithmetic intensity: 262 KFLOP vs 8 KB L2 traffic per wave-step = 32 FLOP/B.
__global__ __launch_bounds__(256) void bl_gemm_kernel(const _Float16* __restrict__ xq,
                                                      const _Float16* __restrict__ wq_t,
                                                      const float* __restrict__ gamma,
                                                      const unsigned* __restrict__ scalars,
                                                      float* __restrict__ out) {
    const int lane  = threadIdx.x & 31;
    const int wave  = threadIdx.x >> 5;        // 0..7
    const int waveM = wave >> 2;               // 0..1
    const int waveN = wave & 3;                // 0..3
    const int rowBase = blockIdx.y * 128 + waveM * 64;
    const int colBase = blockIdx.x * 256 + waveN * 64;
    const int l16   = lane & 15;
    const int khalf = (lane >> 4) << 3;        // 0 or 8

    const _Float16* aBase = xq   + (size_t)(rowBase + l16) * K_DIM + khalf;
    const _Float16* bBase = wq_t + (size_t)(colBase + l16) * K_DIM + khalf;

    v8f c[4][4] = {};

    for (int k0 = 0; k0 < K_DIM; k0 += 32) {
        v16h fa[4], fb[4];
#pragma unroll
        for (int i = 0; i < 4; ++i) fa[i] = bl_load_frag(aBase + (size_t)i * 16 * K_DIM + k0);
#pragma unroll
        for (int i = 0; i < 4; ++i) fb[i] = bl_load_frag(bBase + (size_t)i * 16 * K_DIM + k0);
#pragma unroll
        for (int mt = 0; mt < 4; ++mt)
#pragma unroll
            for (int nt = 0; nt < 4; ++nt)
                c[mt][nt] = __builtin_amdgcn_wmma_f32_16x16x32_f16(
                    false, fa[mt], false, fb[nt], (short)0, c[mt][nt], false, false);
    }

    const float beta = __uint_as_float(*(const volatile unsigned*)&scalars[1]);
    // C/D layout: VGPR r, lanes 0-15 -> (M=r, N=lane); lanes 16-31 -> (M=r+8, N=lane-16)
    const int mOff = (lane >> 4) << 3;         // +8 for upper lane half
#pragma unroll
    for (int mt = 0; mt < 4; ++mt) {
#pragma unroll
        for (int r = 0; r < 8; ++r) {
            const int m = rowBase + mt * 16 + mOff + r;
            const float sc = beta * gamma[m];   // QB == 1
#pragma unroll
            for (int nt = 0; nt < 4; ++nt) {
                const int n = colBase + nt * 16 + l16;
                out[(size_t)m * N_OUT + n] = c[mt][nt][r] * sc;
            }
        }
    }
}

// ---------------------------------------------------------------------------
extern "C" void kernel_launch(void* const* d_in, const int* in_sizes, int n_in,
                              void* d_out, int out_size, void* d_ws, size_t ws_size,
                              hipStream_t stream) {
    const float* x = (const float*)d_in[0];        // [32768, 1024] f32
    const float* w = (const float*)d_in[1];        // [1024, 1024]  f32
    float* out = (float*)d_out;                    // [32768, 1024] f32

    char* ws = (char*)d_ws;
    unsigned* scalars = (unsigned*)(ws + WS_SCALARS);
    float*    gamma   = (float*)(ws + WS_GAMMA);
    _Float16* xq      = (_Float16*)(ws + WS_XQ);
    _Float16* wq_t    = (_Float16*)(ws + WS_WQ);

    (void)in_sizes; (void)n_in; (void)out_size; (void)ws_size;

    bl_init_kernel<<<1, 32, 0, stream>>>(scalars);

    bl_wreduce_kernel<<<(K_DIM * N_OUT) / (256 * 4), 256, 0, stream>>>(w, scalars);

    dim3 wq_grid(N_OUT / 32, K_DIM / 32);
    bl_wquant_kernel<<<wq_grid, 256, 0, stream>>>(w, scalars, wq_t);

    bl_xquant_kernel<<<N_ROWS, 256, 0, stream>>>(x, gamma, xq);

    dim3 g_grid(N_OUT / 256, N_ROWS / 128);
    bl_gemm_kernel<<<g_grid, 256, 0, stream>>>(xq, wq_t, gamma, scalars, out);
}
